// MiniCPMV4_6ViTWindowAttentionMerger_84507776516289
// MI455X (gfx1250) — matrix operand order repaired and checked by reading, not compiled, not run on hardware
//
#include <hip/hip_runtime.h>
#include <stdint.h>

// ---------------------------------------------------------------------------
// MiniCPMV4.6 ViT window-attention + 2x2 merger, CDNA5 (gfx1250) WMMA version.
// All heavy GEMMs use v_wmma_f32_16x16x32_bf16; activations kept in bf16.
// ---------------------------------------------------------------------------

typedef unsigned short ushort_t;
typedef __attribute__((ext_vector_type(16))) __bf16 v16bf;
typedef __attribute__((ext_vector_type(8)))  float  v8f;

union FragBF { uint4 u[2]; v16bf v; };

// native converts: fptrunc f32->bf16 is RNE; compiler can select v_cvt_pk_bf16_f32
__device__ __forceinline__ ushort_t f2bf(float f) {
  union { __bf16 h; ushort_t u; } v; v.h = (__bf16)f; return v.u;
}
__device__ __forceinline__ unsigned f2bf_pk(float lo, float hi) {
  union { __bf16 h[2]; unsigned u; } v;
  v.h[0] = (__bf16)lo; v.h[1] = (__bf16)hi; return v.u;
}
__device__ __forceinline__ float bf2f(ushort_t h) {
  union { unsigned u; float f; } v; v.u = ((unsigned)h) << 16;
  return v.f;
}

// Problem constants
#define NB      8
#define HW      4096          // 64*64 tokens per image
#define TOK     32768         // NB*HW
#define DH      1152
#define QKVN    3456
#define NWIN    8192          // NB * 32*32 windows
#define H4      4608          // 4*DH
#define I4      17216
#define NHID    9437184       // NWIN * DH (new_hidden elements)

// ---------------------------------------------------------------------------
// GEMM: C[M,N] = A(bf16)[M,K] x B(f32->bf16)[K,N] + bias (+GELU) (+res)
// Block tile 256x64, 8 waves as 4(M)x2(N), each wave 64x32 = 4x2 WMMA tiles.
// K-step 32, register-prefetch software pipeline over global loads.
// ---------------------------------------------------------------------------
template<bool GELU, bool RES, bool OBF16>
__global__ __launch_bounds__(256) void gemm_bf16_wmma(
    const ushort_t* __restrict__ A, int lda,
    const float*    __restrict__ Bw, int ldb,
    const float*    __restrict__ bias,
    const float*    __restrict__ res,
    void*           __restrict__ Cout, int ldc,
    int K)
{
  __shared__ __align__(16) ushort_t lsA[256 * 40];  // 256 rows x 32 bf16, pad 8
  __shared__ __align__(16) ushort_t lsB[64 * 40];   // 64 cols  x 32 bf16 (B^T)

  const int tid  = threadIdx.x;
  const int lane = tid & 31;
  const int wave = tid >> 5;
  const int wm   = wave >> 1;            // 0..3  (M direction, 64 rows each)
  const int wn   = wave & 1;             // 0..1  (N direction, 32 cols each)
  const int m0   = blockIdx.y * 256;
  const int n0   = blockIdx.x * 64;
  const int l15  = lane & 15;
  const int hiA  = (lane & 16) ? 8 : 0;  // ISA A-layout K sub-offset (ushorts)
  const int hiB  = (lane & 16) ? 8 : 0;  // B fragment K offset (ushorts)

  // per-thread staging coordinates (hoisted pointers, constant strides)
  const int arow = tid >> 2;             // 0..63 (+64/+128/+192 for other chunks)
  const int akc  = (tid & 3) << 3;       // 0,8,16,24
  const ushort_t* pA0 = A + (size_t)(m0 + arow) * lda + akc;        // rows 0..63
  const ushort_t* pA1 = pA0 + (size_t)64  * lda;                    // rows 64..127
  const ushort_t* pA2 = pA0 + (size_t)128 * lda;                    // rows 128..191
  const ushort_t* pA3 = pA0 + (size_t)192 * lda;                    // rows 192..255
  const int kk = (tid >> 4) << 1;        // 0,2,..,30
  const int nn = (tid & 15) << 2;        // 0,4,..,60
  const float* pB0 = Bw + (size_t)kk * ldb + n0 + nn;
  const float* pB1 = pB0 + ldb;
  const size_t bstep = (size_t)32 * ldb;

  v8f acc[4][2] = {};

  // pipeline prologue: fetch k-tile 0 into registers
  uint4  ra0 = *(const uint4*)pA0;
  uint4  ra1 = *(const uint4*)pA1;
  uint4  ra2 = *(const uint4*)pA2;
  uint4  ra3 = *(const uint4*)pA3;
  float4 rb0 = *(const float4*)pB0;
  float4 rb1 = *(const float4*)pB1;

  for (int k0 = 0; k0 < K; k0 += 32) {
    // commit staged registers to LDS (B: packed bf16 pairs via v_cvt_pk)
    *(uint4*)&lsA[(arow      ) * 40 + akc] = ra0;
    *(uint4*)&lsA[(arow +  64) * 40 + akc] = ra1;
    *(uint4*)&lsA[(arow + 128) * 40 + akc] = ra2;
    *(uint4*)&lsA[(arow + 192) * 40 + akc] = ra3;
    #pragma unroll
    for (int j = 0; j < 4; ++j) {
      *(unsigned*)&lsB[(nn + j) * 40 + kk] =
          f2bf_pk(((const float*)&rb0)[j], ((const float*)&rb1)[j]);
    }
    __syncthreads();

    // prefetch next k-tile into registers (overlaps with WMMA below)
    if (k0 + 32 < K) {
      pA0 += 32; pA1 += 32; pA2 += 32; pA3 += 32;
      pB0 += bstep; pB1 += bstep;
      ra0 = *(const uint4*)pA0;
      ra1 = *(const uint4*)pA1;
      ra2 = *(const uint4*)pA2;
      ra3 = *(const uint4*)pA3;
      rb0 = *(const float4*)pB0;
      rb1 = *(const float4*)pB1;
      __builtin_prefetch(pB0 + 8 * bstep, 0, 1);   // weight stream, 8 tiles ahead
    }

    FragBF a[4], b[2];
    #pragma unroll
    for (int tm = 0; tm < 4; ++tm) {
      int row = wm * 64 + tm * 16 + l15;
      const uint4* p = (const uint4*)&lsA[row * 40 + hiA];
      a[tm].u[0] = p[0];            // K runs {0..7} / {8..15}
      a[tm].u[1] = p[2];            // K runs {16..23} / {24..31}
    }
    #pragma unroll
    for (int tn = 0; tn < 2; ++tn) {
      int col = wn * 32 + tn * 16 + l15;
      const uint4* p = (const uint4*)&lsB[col * 40 + hiB];
      b[tn].u[0] = p[0];            // K runs {0..7} / {8..15}
      b[tn].u[1] = p[2];            // K runs {16..23} / {24..31}
    }
    #pragma unroll
    for (int tm = 0; tm < 4; ++tm)
      #pragma unroll
      for (int tn = 0; tn < 2; ++tn)
        acc[tm][tn] = __builtin_amdgcn_wmma_f32_16x16x32_bf16(
            false, a[tm].v, false, b[tn].v, (short)0, acc[tm][tn], false, false);
    __syncthreads();
  }

  // epilogue: bias (+GELU) (+residual), C layout per ISA 16x16 f32 D matrix
  #pragma unroll
  for (int tn = 0; tn < 2; ++tn) {
    const int gn = n0 + wn * 32 + tn * 16 + l15;
    const float bv = bias[gn];
    #pragma unroll
    for (int tm = 0; tm < 4; ++tm) {
      #pragma unroll
      for (int r = 0; r < 8; ++r) {
        const int gm = m0 + wm * 64 + tm * 16 + r + ((lane & 16) ? 8 : 0);
        float v = acc[tm][tn][r] + bv;
        if (GELU) {
          float x3 = v * v * v;
          v = 0.5f * v * (1.0f + tanhf(0.7978845608f * (v + 0.044715f * x3)));
        }
        if (RES) v += res[(size_t)gm * ldc + gn];
        if (OBF16) ((ushort_t*)Cout)[(size_t)gm * ldc + gn] = f2bf(v);
        else       ((float*)Cout)[(size_t)gm * ldc + gn]    = v;
      }
    }
  }
}

// ---------------------------------------------------------------------------
// LayerNorm over D=1152, one wave per token, wave32 shuffle reduction.
// ---------------------------------------------------------------------------
__global__ __launch_bounds__(256) void ln1_kernel(
    const float* __restrict__ x, const float* __restrict__ g,
    const float* __restrict__ bt, ushort_t* __restrict__ y)
{
  const int wave = (blockIdx.x * blockDim.x + threadIdx.x) >> 5;
  const int lane = threadIdx.x & 31;
  const float* row = x + (size_t)wave * DH;
  float v[36], s = 0.f, ss = 0.f;
  #pragma unroll
  for (int j = 0; j < 36; ++j) {
    float t = row[lane + j * 32];
    v[j] = t; s += t; ss += t * t;
  }
  #pragma unroll
  for (int o = 16; o > 0; o >>= 1) {
    s  += __shfl_xor(s,  o, 32);
    ss += __shfl_xor(ss, o, 32);
  }
  const float mean = s * (1.0f / DH);
  const float rstd = rsqrtf(ss * (1.0f / DH) - mean * mean + 1e-6f);
  ushort_t* yr = y + (size_t)wave * DH;
  #pragma unroll
  for (int j = 0; j < 36; ++j) {
    int idx = lane + j * 32;
    yr[idx] = f2bf((v[j] - mean) * rstd * g[idx] + bt[idx]);
  }
}

// ---------------------------------------------------------------------------
// 4-token window attention, 16 heads, hd=72. One thread per (window,head,q).
// ---------------------------------------------------------------------------
__global__ __launch_bounds__(256) void attn_kernel(
    const ushort_t* __restrict__ qkv, ushort_t* __restrict__ o)
{
  const int t = blockIdx.x * blockDim.x + threadIdx.x;
  if (t >= NWIN * 16 * 4) return;
  const int qi   = t & 3;
  const int head = (t >> 2) & 15;
  const int w    = t >> 6;
  const int b = w >> 10, wi = w & 1023, ih = wi >> 5, iw = wi & 31;
  int rows[4];
  #pragma unroll
  for (int r = 0; r < 2; ++r)
    #pragma unroll
    for (int c = 0; c < 2; ++c)
      rows[r * 2 + c] = (b << 12) + (2 * ih + r) * 64 + (2 * iw + c);

  const int qoff = head * 72, koff = DH + head * 72, voff = 2 * DH + head * 72;
  const ushort_t* qrow = qkv + (size_t)rows[qi] * QKVN + qoff;
  const float scale = 0.11785113019775793f;   // 72^-0.5
  float sc[4];
  #pragma unroll
  for (int j = 0; j < 4; ++j) {
    const ushort_t* krow = qkv + (size_t)rows[j] * QKVN + koff;
    float s = 0.f;
    for (int d = 0; d < 72; ++d) s += bf2f(qrow[d]) * bf2f(krow[d]);
    sc[j] = s * scale;
  }
  float m = fmaxf(fmaxf(sc[0], sc[1]), fmaxf(sc[2], sc[3]));
  float p[4], sum = 0.f;
  #pragma unroll
  for (int j = 0; j < 4; ++j) { p[j] = __expf(sc[j] - m); sum += p[j]; }
  const float inv = 1.0f / sum;
  #pragma unroll
  for (int j = 0; j < 4; ++j) p[j] *= inv;

  ushort_t* orow = o + (size_t)rows[qi] * DH + head * 72;
  for (int d = 0; d < 72; ++d) {
    float a = 0.f;
    #pragma unroll
    for (int j = 0; j < 4; ++j)
      a += p[j] * bf2f(qkv[(size_t)rows[j] * QKVN + voff + d]);
    orow[d] = f2bf(a);
  }
}

// ---------------------------------------------------------------------------
// Merge 2x2 window: mean residual + LayerNorm over 4608, one block per window.
// ---------------------------------------------------------------------------
__global__ __launch_bounds__(256) void merge_ln_kernel(
    const float* __restrict__ hs, const float* __restrict__ g,
    const float* __restrict__ bt, ushort_t* __restrict__ xln,
    float* __restrict__ resm)
{
  __shared__ float shs[256], shss[256];
  const int w = blockIdx.x;
  const int b = w >> 10, wi = w & 1023, ih = wi >> 5, iw = wi & 31;
  int rows[4];
  #pragma unroll
  for (int r = 0; r < 2; ++r)
    #pragma unroll
    for (int c = 0; c < 2; ++c)
      rows[r * 2 + c] = (b << 12) + (2 * ih + r) * 64 + (2 * iw + c);

  const int tid = threadIdx.x;
  float s = 0.f, ss = 0.f;
  for (int e = tid; e < H4; e += 256) {
    int tt = e / DH, d = e - tt * DH;
    float x = hs[(size_t)rows[tt] * DH + d];
    s += x; ss += x * x;
  }
  shs[tid] = s; shss[tid] = ss;
  __syncthreads();
  for (int o = 128; o > 0; o >>= 1) {
    if (tid < o) { shs[tid] += shs[tid + o]; shss[tid] += shss[tid + o]; }
    __syncthreads();
  }
  const float mean = shs[0] * (1.0f / H4);
  const float rstd = rsqrtf(shss[0] * (1.0f / H4) - mean * mean + 1e-6f);

  ushort_t* xr = xln + (size_t)w * H4;
  for (int e = tid; e < H4; e += 256) {
    int tt = e / DH, d = e - tt * DH;
    float x = hs[(size_t)rows[tt] * DH + d];
    xr[e] = f2bf((x - mean) * rstd * g[e] + bt[e]);
  }
  for (int d = tid; d < DH; d += 256) {
    float m4 = hs[(size_t)rows[0] * DH + d] + hs[(size_t)rows[1] * DH + d] +
               hs[(size_t)rows[2] * DH + d] + hs[(size_t)rows[3] * DH + d];
    resm[(size_t)w * DH + d] = 0.25f * m4;
  }
}

// ---------------------------------------------------------------------------
// Tail: tgt_sizes // 2 (as int bits) and zero attention mask.
// ---------------------------------------------------------------------------
__global__ __launch_bounds__(256) void tail_kernel(
    const int* __restrict__ tgt, float* __restrict__ out)
{
  const int tid = blockIdx.x * blockDim.x + threadIdx.x;
  if (blockIdx.x == 0 && threadIdx.x < 16)
    ((int*)out)[NHID + threadIdx.x] = tgt[threadIdx.x] >> 1;
  for (int i = tid; i < NWIN; i += gridDim.x * blockDim.x)
    out[NHID + 16 + i] = 0.0f;   // all windows valid -> mask = 0
}

// ---------------------------------------------------------------------------
extern "C" void kernel_launch(void* const* d_in, const int* in_sizes, int n_in,
                              void* d_out, int out_size, void* d_ws, size_t ws_size,
                              hipStream_t stream)
{
  (void)in_sizes; (void)n_in; (void)out_size; (void)ws_size;
  const float* hidden = (const float*)d_in[0];
  const int*   tgt    = (const int*)d_in[1];
  // d_in[2]: attention_mask (zeros, unused)
  const float* ln1_g = (const float*)d_in[3];
  const float* ln1_b = (const float*)d_in[4];
  const float* qkv_w = (const float*)d_in[5];
  const float* qkv_b = (const float*)d_in[6];
  const float* out_w = (const float*)d_in[7];
  const float* out_b = (const float*)d_in[8];
  const float* pre_g = (const float*)d_in[9];
  const float* pre_b = (const float*)d_in[10];
  const float* w1    = (const float*)d_in[11];
  const float* b1    = (const float*)d_in[12];
  const float* w2    = (const float*)d_in[13];
  const float* b2    = (const float*)d_in[14];

  char* ws = (char*)d_ws;
  size_t off = 0;
  auto grab = [&](size_t bytes) { char* p = ws + off; off += (bytes + 255) & ~size_t(255); return p; };
  ushort_t* ln_out = (ushort_t*)grab((size_t)TOK * DH * 2);    // bf16 LN(hidden)
  ushort_t* qkv    = (ushort_t*)grab((size_t)TOK * QKVN * 2);  // bf16 qkv
  ushort_t* o      = (ushort_t*)grab((size_t)TOK * DH * 2);    // bf16 attn ctx
  float*    hs     = (float*)   grab((size_t)TOK * DH * 4);    // f32 post-attn
  float*    resm   = (float*)   grab((size_t)NWIN * DH * 4);   // f32 mean res
  ushort_t* xln    = (ushort_t*)grab((size_t)NWIN * H4 * 2);   // bf16 LN(merge)
  ushort_t* h1     = (ushort_t*)grab((size_t)NWIN * I4 * 2);   // bf16 gelu(mlp1)

  // 1) LN1 (wave per token): 32768 tokens / 8 waves-per-block
  ln1_kernel<<<TOK / 8, 256, 0, stream>>>(hidden, ln1_g, ln1_b, ln_out);

  // 2) QKV GEMM: [32768,1152] x [1152,3456] + bias -> bf16
  gemm_bf16_wmma<false, false, true><<<dim3(QKVN / 64, TOK / 256), 256, 0, stream>>>(
      ln_out, DH, qkv_w, QKVN, qkv_b, nullptr, qkv, QKVN, DH);

  // 3) window attention (VALU; tiny)
  attn_kernel<<<(NWIN * 16 * 4) / 256, 256, 0, stream>>>(qkv, o);

  // 4) out-proj GEMM: [32768,1152] x [1152,1152] + bias + residual -> f32 hs
  gemm_bf16_wmma<false, true, false><<<dim3(DH / 64, TOK / 256), 256, 0, stream>>>(
      o, DH, out_w, DH, out_b, hidden, hs, DH, DH);

  // 5) merge 2x2 + LN(4608) + mean residual
  merge_ln_kernel<<<NWIN, 256, 0, stream>>>(hs, pre_g, pre_b, xln, resm);

  // 6) MLP1: [8192,4608] x [4608,17216] + bias + GELU -> bf16
  gemm_bf16_wmma<true, false, true><<<dim3(I4 / 64, NWIN / 256), 256, 0, stream>>>(
      xln, H4, w1, I4, b1, nullptr, h1, I4, H4);

  // 7) MLP2: [8192,17216] x [17216,1152] + bias + mean-res -> d_out (f32)
  gemm_bf16_wmma<false, true, false><<<dim3(DH / 64, NWIN / 256), 256, 0, stream>>>(
      h1, I4, w2, DH, b2, resm, (float*)d_out, DH, I4);

  // 8) tgt_sizes // 2 and zero mask
  tail_kernel<<<32, 256, 0, stream>>>(tgt, (float*)d_out);
}
